// GraphAttentionLayer_69501160784119
// MI455X (gfx1250) — matrix operand order repaired and checked by reading, not compiled, not run on hardware
//
#include <hip/hip_runtime.h>
#include <hip/hip_bf16.h>

typedef __attribute__((ext_vector_type(16))) _Float16 v16h;
typedef __attribute__((ext_vector_type(8)))  float    v8f;

#define LEAKY_ALPHA 0.2f
#define B_  4
#define N_  4096
#define F_  128
#define U_  64

// ---------------------------------------------------------------------------
// Kernel 1: h = x @ W  (per row), f1 = h.a1, f2 = h.a2, and hT (f16, [B][U][N])
// One 64-thread block per (b,n) row. 268 MFLOP total -> scalar f32 FMA is fine
// and keeps f1/f2 (softmax logits) in full precision.
// ---------------------------------------------------------------------------
__global__ __launch_bounds__(64) void gat_h_kernel(
    const float* __restrict__ x,   // [B][N][F]
    const float* __restrict__ Wm,  // [F][U]
    const float* __restrict__ av,  // [2U]
    _Float16* __restrict__ hT,     // [B][U][N]
    float* __restrict__ f1,        // [B*N]
    float* __restrict__ f2)        // [B*N]
{
    int gn = blockIdx.x;              // 0 .. B*N-1
    int b  = gn >> 12;
    int n  = gn & (N_ - 1);
    int t  = threadIdx.x;             // 0..63 == u

    __shared__ float xs[F_];
    __shared__ float hs[U_];

    const float* xr = x + (size_t)gn * F_;
    xs[t]      = xr[t];
    xs[t + 64] = xr[t + 64];
    __syncthreads();

    float acc = 0.f;
#pragma unroll 8
    for (int f = 0; f < F_; ++f)
        acc = fmaf(xs[f], Wm[f * U_ + t], acc);   // W read coalesced across u

    hs[t] = acc;
    hT[((size_t)b * U_ + t) * N_ + n] = (_Float16)acc;
    __syncthreads();

    if (t < 2) {
        const float* ap = av + t * U_;
        float s = 0.f;
#pragma unroll
        for (int u = 0; u < U_; ++u) s = fmaf(hs[u], ap[u], s);
        (t == 0 ? f1 : f2)[gn] = s;
    }
}

// ---------------------------------------------------------------------------
// Kernel 2: maxf2[b] = max_j f2[b][j]  (exact row-max enabler: lrelu monotone)
// ---------------------------------------------------------------------------
__global__ __launch_bounds__(128) void gat_max_kernel(
    const float* __restrict__ f2, float* __restrict__ maxf2)
{
    int b = blockIdx.x;
    int t = threadIdx.x;
    const float* p = f2 + b * N_;
    float m = -3.402823466e+38f;
    for (int i = t; i < N_; i += 128) m = fmaxf(m, p[i]);
    __shared__ float sm[128];
    sm[t] = m;
    __syncthreads();
    for (int s = 64; s > 0; s >>= 1) {
        if (t < s) sm[t] = fmaxf(sm[t], sm[t + s]);
        __syncthreads();
    }
    if (t == 0) maxf2[b] = sm[0];
}

// ---------------------------------------------------------------------------
// Kernel 3: fused flash-attention.  Block = (batch b, 16-row i-tile), 4 waves.
// Wave w handles j in [w*1024, w*1024+1024) in K=32 chunks; builds the prob
// tile P (16x32, f16) in contiguous-K lane order (k = 16*half + s) and issues 4x
// v_wmma_f32_16x16x32_f16 against hT columns (contiguous v16h loads).
// Denominator uses 4 independent partial accumulators so the VALU adds have a
// short dependency chain and co-execute under the WMMAs.
// Cross-wave combine + softmax normalization via LDS.
// ---------------------------------------------------------------------------
__global__ __launch_bounds__(128) void gat_attn_kernel(
    const _Float16* __restrict__ hT,    // [B][U][N]
    const float* __restrict__ f1,       // [B*N]
    const float* __restrict__ f2,       // [B*N]
    const float* __restrict__ maxf2,    // [B]
    float* __restrict__ out)            // [B][N][U]
{
    int b    = blockIdx.y;
    int i0   = blockIdx.x * 16;
    int tid  = threadIdx.x;
    int w    = tid >> 5;        // wave id 0..3
    int lane = tid & 31;
    int half = lane >> 4;       // 0: K 0-15 ; 1: K 16-31
    int lrow = lane & 15;       // A-matrix row / D-matrix column

    float f1i = f1[b * N_ + i0 + lrow];
    float m   = f1i + maxf2[b];
    m = fmaxf(m, LEAKY_ALPHA * m);          // lrelu(f1_i + max_j f2_j) == exact row max

    v8f acc0 = {}, acc1 = {}, acc2 = {}, acc3 = {};
    float den0 = 0.f, den1 = 0.f, den2 = 0.f, den3 = 0.f;

    const float*     f2b = f2 + b * N_;
    const _Float16*  hTb = hT + (size_t)b * U_ * N_;

    int jbeg = w * (N_ / 4);
    int jend = jbeg + (N_ / 4);
#pragma unroll 2
    for (int j0 = jbeg; j0 < jend; j0 += 32) {
        // per-lane f2 slice in contiguous-K order: element s is k = 16*half + s
        int base = j0 + 16 * half;
        float4 q0 = *(const float4*)(f2b + base);
        float4 q1 = *(const float4*)(f2b + base + 4);
        float4 q2 = *(const float4*)(f2b + base + 8);
        float4 q3 = *(const float4*)(f2b + base + 12);
        float fv[16] = {q0.x, q0.y, q0.z, q0.w, q1.x, q1.y, q1.z, q1.w,
                        q2.x, q2.y, q2.z, q2.w, q3.x, q3.y, q3.z, q3.w};
        float pv[16];
        v16h a;
#pragma unroll
        for (int k = 0; k < 16; ++k) {
            float s = f1i + fv[k];
            s = fmaxf(s, LEAKY_ALPHA * s);      // leaky relu, branchless
            float p = __expf(s - m);            // v_exp_f32 (TRANS)
            pv[k] = p;
            a[k] = (_Float16)p;
        }
        // 4 independent partial sums -> dependency chain of 4, not 16
#pragma unroll
        for (int k = 0; k < 4; ++k) {
            den0 += pv[4 * k + 0];
            den1 += pv[4 * k + 1];
            den2 += pv[4 * k + 2];
            den3 += pv[4 * k + 3];
        }

        // B tiles: hT columns, K contiguous -> two global_load_b128 each
        int koff = j0 + 16 * half;
        const _Float16* colbase = hTb + koff;
        v16h b0 = *(const v16h*)(colbase + (size_t)(lrow)      * N_);
        v16h b1 = *(const v16h*)(colbase + (size_t)(16 + lrow) * N_);
        v16h b2 = *(const v16h*)(colbase + (size_t)(32 + lrow) * N_);
        v16h b3 = *(const v16h*)(colbase + (size_t)(48 + lrow) * N_);

        acc0 = __builtin_amdgcn_wmma_f32_16x16x32_f16(false, a, false, b0, (short)0, acc0, false, false);
        acc1 = __builtin_amdgcn_wmma_f32_16x16x32_f16(false, a, false, b1, (short)0, acc1, false, false);
        acc2 = __builtin_amdgcn_wmma_f32_16x16x32_f16(false, a, false, b2, (short)0, acc2, false, false);
        acc3 = __builtin_amdgcn_wmma_f32_16x16x32_f16(false, a, false, b3, (short)0, acc3, false, false);
    }
    float den = (den0 + den1) + (den2 + den3);

    // ---- cross-wave reduction ----
    __shared__ float red[4][16][64];   // 16 KB
    __shared__ float denp[4][32];
    __shared__ float denr[16];

#pragma unroll
    for (int v = 0; v < 8; ++v) {
        int M = v + 8 * half;          // D layout: lanes 0-15 -> M=v, 16-31 -> M=v+8
        red[w][M][ 0 + lrow] = acc0[v];
        red[w][M][16 + lrow] = acc1[v];
        red[w][M][32 + lrow] = acc2[v];
        red[w][M][48 + lrow] = acc3[v];
    }
    denp[w][lane] = den;
    __syncthreads();

    if (tid < 16) {
        float s = 0.f;
#pragma unroll
        for (int ww = 0; ww < 4; ++ww) s += denp[ww][tid] + denp[ww][tid + 16];
        denr[tid] = s;
    }
    __syncthreads();

    int row = tid >> 3;            // 0..15
    int c0  = (tid & 7) * 8;       // 8 consecutive u per thread -> coalesced
    float inv = 1.0f / denr[row];
    float* op = out + (((size_t)b * N_ + i0 + row) * U_) + c0;
#pragma unroll
    for (int c = 0; c < 8; ++c) {
        float s = red[0][row][c0 + c] + red[1][row][c0 + c]
                + red[2][row][c0 + c] + red[3][row][c0 + c];
        op[c] = s * inv;
    }
}

// ---------------------------------------------------------------------------
extern "C" void kernel_launch(void* const* d_in, const int* in_sizes, int n_in,
                              void* d_out, int out_size, void* d_ws, size_t ws_size,
                              hipStream_t stream) {
    const float* x  = (const float*)d_in[0];   // (B,N,F) f32
    const float* Wm = (const float*)d_in[1];   // (F,U)   f32
    const float* av = (const float*)d_in[2];   // (2U,1)  f32
    float* out = (float*)d_out;                // (B,N,U) f32

    char* ws = (char*)d_ws;
    const size_t hT_bytes = (size_t)B_ * U_ * N_ * sizeof(_Float16); // 2 MB
    const size_t fx_bytes = (size_t)B_ * N_ * sizeof(float);         // 64 KB
    _Float16* hT = (_Float16*)ws;
    float* f1 = (float*)(ws + hT_bytes);
    float* f2 = (float*)(ws + hT_bytes + fx_bytes);
    float* mf = (float*)(ws + hT_bytes + 2 * fx_bytes);

    gat_h_kernel   <<<dim3(B_ * N_),   64, 0, stream>>>(x, Wm, av, hT, f1, f2);
    gat_max_kernel <<<dim3(B_),       128, 0, stream>>>(f2, mf);
    gat_attn_kernel<<<dim3(N_/16, B_),128, 0, stream>>>(hT, f1, f2, mf, out);
}
